// OlmoHybridGatedDeltaNet_58342835748958
// MI455X (gfx1250) — matrix operand order repaired and checked by reading, not compiled, not run on hardware
//
#include <hip/hip_runtime.h>
#include <hip/hip_bf16.h>

// ---------------------------------------------------------------------------
// Problem constants (from reference)
// ---------------------------------------------------------------------------
#define LSEQ 4096
#define HDIM 2048
#define HK 8
#define HV 16
#define DK 64
#define DV 64
#define KCONV 4
#define KEY_DIM   (HK * DK)                     // 512
#define VALUE_DIM (HV * DV)                     // 1024
#define CONV_DIM  (2 * KEY_DIM + VALUE_DIM)     // 2048
#define PROJ_DIM  (2 * KEY_DIM + 2 * VALUE_DIM) // 3072

typedef __bf16 v16bf __attribute__((ext_vector_type(16)));
typedef __bf16 v8bf  __attribute__((ext_vector_type(8)));
typedef float  v8f   __attribute__((ext_vector_type(8)));
typedef int    v4i   __attribute__((ext_vector_type(4)));

// ---- gfx1250 async global->LDS copy (ASYNCcnt path), with fallbacks --------
#if defined(__has_builtin)
#  if __has_builtin(__builtin_amdgcn_global_load_async_to_lds_b128)
#    define HAVE_ASYNC_LDS 1
#  else
#    define HAVE_ASYNC_LDS 0
#  endif
#  if __has_builtin(__builtin_amdgcn_s_wait_asynccnt)
#    define HAVE_WAIT_ASYNC 1
#  else
#    define HAVE_WAIT_ASYNC 0
#  endif
#else
#  define HAVE_ASYNC_LDS 0
#  define HAVE_WAIT_ASYNC 0
#endif

// Builtin parameter types (from hipcc diagnostic): int4 pointers in
// global (AS1) / LDS (AS3) address spaces.
typedef __attribute__((address_space(1))) v4i gv4i;
typedef __attribute__((address_space(3))) v4i lv4i;

__device__ __forceinline__ void copy16_g2l(const void* g, void* l) {
#if HAVE_ASYNC_LDS
    __builtin_amdgcn_global_load_async_to_lds_b128(
        (gv4i*)(uintptr_t)g, (lv4i*)(uintptr_t)l, 0, 0);
#else
    *(uint4*)l = *(const uint4*)g;   // global_load_b128 + ds_store_b128
#endif
}

__device__ __forceinline__ void async_copy_wait() {
#if HAVE_ASYNC_LDS
#  if HAVE_WAIT_ASYNC
    __builtin_amdgcn_s_wait_asynccnt(0);
#  else
    asm volatile("s_wait_asynccnt 0x0" ::: "memory");
#  endif
#endif
}

__device__ __forceinline__ __bf16 f2bf(float f) {
    union { float f; unsigned u; } x; x.f = f;
    unsigned r = (x.u + 0x7fffu + ((x.u >> 16) & 1u)) >> 16;  // RNE
    union { unsigned short s; __bf16 b; } y; y.s = (unsigned short)r;
    return y.b;
}

// ---------------------------------------------------------------------------
// One-time f32 -> bf16 conversion (amortized over all GEMM tile reuses)
// ---------------------------------------------------------------------------
__global__ __launch_bounds__(256)
void f32_to_bf16_kernel(const float* __restrict__ src, __bf16* __restrict__ dst) {
    size_t i = ((size_t)blockIdx.x * 256 + threadIdx.x) * 4;
    float4 f = *(const float4*)(src + i);
    __bf16 h[4] = { f2bf(f.x), f2bf(f.y), f2bf(f.z), f2bf(f.w) };
    *(uint2*)(dst + i) = *(const uint2*)h;
}

// ---------------------------------------------------------------------------
// bf16 WMMA GEMM:  C[M,N] = A[M,K] (bf16) * W[N,K]^T (bf16), f32 accumulate.
// 256 threads (8 wave32), block tile 128(M) x 256(N), K-step 32.
// Double-buffered LDS; async global->LDS copies overlap the 16 WMMAs/step.
// ---------------------------------------------------------------------------
__global__ __launch_bounds__(256)
void gemm_bf16_wmma(const __bf16* __restrict__ A,
                    const __bf16* __restrict__ W,
                    float* __restrict__ C,
                    int M, int N, int K) {
    __shared__ __align__(16) __bf16 As[2][128 * 32];   // [m][k]
    __shared__ __align__(16) __bf16 Bs[2][256 * 32];   // [n][k]

    const int tid  = threadIdx.x;
    const int lane = tid & 31;
    const int wv   = tid >> 5;            // 0..7
    const int waveM = wv >> 2;            // 0..1  -> 64-row slab
    const int waveN = wv & 3;             // 0..3  -> 64-col slab

    const int blockM = blockIdx.y * 128;
    const int blockN = blockIdx.x * 256;

    v8f acc[4][4];
#pragma unroll
    for (int mi = 0; mi < 4; ++mi)
#pragma unroll
        for (int ni = 0; ni < 4; ++ni)
            acc[mi][ni] = (v8f)(0.0f);

    const int ksteps = K >> 5;

    // Stage one 128x32 A tile + 256x32 B tile (bf16) into LDS buffer `buf`.
    auto stage = [&](int buf, int kt) {
        const int kb = kt << 5;
        // A: 512 16-byte chunks, 2 per thread
#pragma unroll
        for (int i = 0; i < 2; ++i) {
            int c = tid + i * 256;
            int row = c >> 2, k8 = (c & 3) * 8;
            copy16_g2l(A + (size_t)(blockM + row) * K + kb + k8,
                       &As[buf][row * 32 + k8]);
        }
        // B: 1024 16-byte chunks, 4 per thread
#pragma unroll
        for (int i = 0; i < 4; ++i) {
            int c = tid + i * 256;
            int n = c >> 2, k8 = (c & 3) * 8;
            copy16_g2l(W + (size_t)(blockN + n) * K + kb + k8,
                       &Bs[buf][n * 32 + k8]);
        }
    };

    stage(0, 0);
    for (int kt = 0; kt < ksteps; ++kt) {
        const int buf = kt & 1;
        async_copy_wait();
        __syncthreads();
        if (kt + 1 < ksteps) stage(buf ^ 1, kt + 1);

        // A fragments: lanes 0-15 hold K 0..7 & 16..23, lanes 16-31 hold 8..15 & 24..31
        v16bf af[4];
        const int akb = (lane < 16) ? 0 : 8;
#pragma unroll
        for (int mi = 0; mi < 4; ++mi) {
            int r = waveM * 64 + mi * 16 + (lane & 15);
            v8bf lo = *(const v8bf*)&As[buf][r * 32 + akb];
            v8bf hi = *(const v8bf*)&As[buf][r * 32 + akb + 16];
#pragma unroll
            for (int j = 0; j < 8; ++j) { af[mi][j] = lo[j]; af[mi][8 + j] = hi[j]; }
        }
        const int bkb = (lane < 16) ? 0 : 8;
#pragma unroll
        for (int ni = 0; ni < 4; ++ni) {
            int n = waveN * 64 + ni * 16 + (lane & 15);
            v8bf blo = *(const v8bf*)&Bs[buf][n * 32 + bkb];
            v8bf bhi = *(const v8bf*)&Bs[buf][n * 32 + bkb + 16];
            v16bf bfrag;
#pragma unroll
            for (int j = 0; j < 8; ++j) { bfrag[j] = blo[j]; bfrag[8 + j] = bhi[j]; }
#pragma unroll
            for (int mi = 0; mi < 4; ++mi) {
                acc[mi][ni] = __builtin_amdgcn_wmma_f32_16x16x32_bf16(
                    false, af[mi], false, bfrag, (short)0, acc[mi][ni], false, false);
            }
        }
    }

    // D 16x16 f32 layout: VGPR j -> M = j (+8 for lanes 16..31), N = lane&15
    const int rsel = (lane < 16) ? 0 : 8;
#pragma unroll
    for (int mi = 0; mi < 4; ++mi) {
#pragma unroll
        for (int ni = 0; ni < 4; ++ni) {
            int row0 = blockM + waveM * 64 + mi * 16 + rsel;
            int col  = blockN + waveN * 64 + ni * 16 + (lane & 15);
#pragma unroll
            for (int j = 0; j < 8; ++j)
                C[(size_t)(row0 + j) * N + col] = acc[mi][ni][j];
        }
    }
}

// ---------------------------------------------------------------------------
// a/b projections: a = hs @ w_a^T, b = hs @ w_b^T  (N=16 each; wave-per-dot)
// ---------------------------------------------------------------------------
__global__ __launch_bounds__(256)
void ab_kernel(const float* __restrict__ hs,
               const float* __restrict__ w_a,
               const float* __restrict__ w_b,
               float* __restrict__ a, float* __restrict__ b) {
    int l = blockIdx.x;
    int wv = threadIdx.x >> 5, lane = threadIdx.x & 31;
    const float* hrow = hs + (size_t)l * HDIM;
#pragma unroll
    for (int j = 0; j < 4; ++j) {
        int o = wv * 4 + j;                         // 0..31
        const float* w = (o < 16) ? (w_a + (size_t)o * HDIM)
                                  : (w_b + (size_t)(o - 16) * HDIM);
        float s = 0.f;
        for (int k = lane; k < HDIM; k += 32) s += hrow[k] * w[k];
#pragma unroll
        for (int off = 16; off; off >>= 1) s += __shfl_xor(s, off, 32);
        if (lane == 0) {
            if (o < 16) a[l * HV + o] = s;
            else        b[l * HV + (o - 16)] = s;
        }
    }
}

// ---------------------------------------------------------------------------
// Causal depthwise conv (KCONV=4) + SiLU over first CONV_DIM cols of proj
// ---------------------------------------------------------------------------
__global__ __launch_bounds__(256)
void conv_silu_kernel(const float* __restrict__ proj,
                      const float* __restrict__ conv_w,
                      float* __restrict__ conv) {
    int idx = blockIdx.x * 256 + threadIdx.x;       // L * CONV_DIM
    int l = idx >> 11, c = idx & (CONV_DIM - 1);
    float s = 0.f;
#pragma unroll
    for (int j = 0; j < KCONV; ++j) {
        int ll = l - (KCONV - 1) + j;
        if (ll >= 0) s += proj[(size_t)ll * PROJ_DIM + c] * conv_w[c * KCONV + j];
    }
    conv[idx] = s / (1.0f + __expf(-s));            // silu
}

// ---------------------------------------------------------------------------
// In-place L2 norm of q/k head rows (q rows also scaled by DK^-0.5)
// ---------------------------------------------------------------------------
__global__ __launch_bounds__(256)
void qknorm_kernel(float* __restrict__ conv) {
    int r = blockIdx.x * 8 + (threadIdx.x >> 5);    // L * 2*HK rows
    int lane = threadIdx.x & 31;
    int l = r >> 4, seg = r & 15;                   // seg 0..7: q, 8..15: k
    float* p = conv + (size_t)l * CONV_DIM + seg * 64;
    float x0 = p[lane * 2], x1 = p[lane * 2 + 1];
    float s = x0 * x0 + x1 * x1;
#pragma unroll
    for (int off = 16; off; off >>= 1) s += __shfl_xor(s, off, 32);
    float sc = rsqrtf(s + 1e-6f);
    if (seg < 8) sc *= 0.125f;                      // DK^-0.5
    p[lane * 2]     = x0 * sc;
    p[lane * 2 + 1] = x1 * sc;
}

// ---------------------------------------------------------------------------
// Gates: eg = exp(-exp(A_log)*softplus(a+dt_bias)), beta = sigmoid(b)
// ---------------------------------------------------------------------------
__global__ __launch_bounds__(256)
void gate_kernel(const float* __restrict__ a, const float* __restrict__ b,
                 const float* __restrict__ A_log, const float* __restrict__ dt_bias,
                 float* __restrict__ eg, float* __restrict__ beta) {
    int idx = blockIdx.x * 256 + threadIdx.x;       // L * HV
    int h = idx & (HV - 1);
    float x  = a[idx] + dt_bias[h];
    float sp = (x > 20.f) ? x : log1pf(__expf(x));
    eg[idx]   = __expf(-__expf(A_log[h]) * sp);
    beta[idx] = 1.0f / (1.0f + __expf(-b[idx]));
}

// ---------------------------------------------------------------------------
// Sequential delta-rule scan. One block per value head (16 blocks, 256 thr).
// State S[64(k)][64(v)] in registers: thread (kg,v) owns S[kg*16..+15][v].
// ---------------------------------------------------------------------------
__global__ __launch_bounds__(256)
void scan_kernel(const float* __restrict__ conv,
                 const float* __restrict__ eg,
                 const float* __restrict__ beta,
                 float* __restrict__ core) {
    const int h  = blockIdx.x;        // 0..HV-1
    const int kh = h >> 1;            // ratio = HV/HK = 2
    const int tid = threadIdx.x;
    const int v  = tid & 63;
    const int kg = tid >> 6;          // 0..3

    __shared__ float ksh[64], qsh[64], red[4][64], upd[64], sc2[2];
    float S[16];
#pragma unroll
    for (int i = 0; i < 16; ++i) S[i] = 0.f;

    for (int l = 0; l < LSEQ; ++l) {
        const float* row = conv + (size_t)l * CONV_DIM;
        if (tid < 64) {
            qsh[tid] = row[kh * 64 + tid];
            ksh[tid] = row[KEY_DIM + kh * 64 + tid];
        }
        if (tid == 0) { sc2[0] = eg[l * HV + h]; sc2[1] = beta[l * HV + h]; }
        __syncthreads();

        const float e = sc2[0], bt = sc2[1];
        float part = 0.f;
#pragma unroll
        for (int i = 0; i < 16; ++i) { S[i] *= e; part += ksh[kg * 16 + i] * S[i]; }
        red[kg][v] = part;
        __syncthreads();

        if (kg == 0) {
            float kv = red[0][v] + red[1][v] + red[2][v] + red[3][v];
            float vt = row[2 * KEY_DIM + h * 64 + v];
            upd[v] = (vt - kv) * bt;
        }
        __syncthreads();

        const float u = upd[v];
        float op = 0.f;
#pragma unroll
        for (int i = 0; i < 16; ++i) { S[i] += ksh[kg * 16 + i] * u; op += qsh[kg * 16 + i] * S[i]; }
        red[kg][v] = op;
        __syncthreads();

        if (kg == 0)
            core[(size_t)l * VALUE_DIM + h * 64 + v] =
                red[0][v] + red[1][v] + red[2][v] + red[3][v];
        __syncthreads();
    }
}

// ---------------------------------------------------------------------------
// RMS-norm + gated SiLU epilogue; emits bf16 directly for the final GEMM.
// ---------------------------------------------------------------------------
__global__ __launch_bounds__(256)
void normgate_kernel(const float* __restrict__ core,
                     const float* __restrict__ proj,
                     const float* __restrict__ o_w,
                     __bf16* __restrict__ normed_bf) {
    int r = blockIdx.x * 8 + (threadIdx.x >> 5);    // L*HV rows
    int lane = threadIdx.x & 31;
    int l = r >> 4, h = r & 15;
    const float* c = core + (size_t)l * VALUE_DIM + h * 64;
    const float* gt = proj + (size_t)l * PROJ_DIM + CONV_DIM + h * 64;
    float x0 = c[lane * 2], x1 = c[lane * 2 + 1];
    float s = x0 * x0 + x1 * x1;
#pragma unroll
    for (int off = 16; off; off >>= 1) s += __shfl_xor(s, off, 32);
    float rs = rsqrtf(s * (1.0f / 64.0f) + 1e-5f);
    float g0 = gt[lane * 2], g1 = gt[lane * 2 + 1];
    float y0 = x0 * rs * o_w[lane * 2]     * (g0 / (1.0f + __expf(-g0)));
    float y1 = x1 * rs * o_w[lane * 2 + 1] * (g1 / (1.0f + __expf(-g1)));
    __bf16 h2[2] = { f2bf(y0), f2bf(y1) };
    *(unsigned*)(normed_bf + (size_t)l * VALUE_DIM + h * 64 + lane * 2) =
        *(const unsigned*)h2;
}

// ---------------------------------------------------------------------------
// Launcher
// ---------------------------------------------------------------------------
extern "C" void kernel_launch(void* const* d_in, const int* in_sizes, int n_in,
                              void* d_out, int out_size, void* d_ws, size_t ws_size,
                              hipStream_t stream) {
    const float* hidden  = (const float*)d_in[0];   // (L, H)
    const float* w_qkvg  = (const float*)d_in[1];   // (3072, 2048)
    const float* w_b     = (const float*)d_in[2];   // (16, 2048)
    const float* w_a     = (const float*)d_in[3];   // (16, 2048)
    const float* conv_w  = (const float*)d_in[4];   // (2048, 4)
    const float* A_log   = (const float*)d_in[5];   // (16,)
    const float* dt_bias = (const float*)d_in[6];   // (16,)
    const float* o_nw    = (const float*)d_in[7];   // (64,)
    const float* w_o     = (const float*)d_in[8];   // (2048, 1024)
    float* out = (float*)d_out;                     // (L, 2048)

    float* ws   = (float*)d_ws;
    float* proj = ws;                                   // L*3072 f32
    float* conv = proj + (size_t)LSEQ * PROJ_DIM;       // L*2048 f32
    float* core = conv + (size_t)LSEQ * CONV_DIM;       // L*1024 f32
    float* aa   = core + (size_t)LSEQ * VALUE_DIM;      // L*16
    float* bb   = aa   + (size_t)LSEQ * HV;
    float* eg   = bb   + (size_t)LSEQ * HV;
    float* bet  = eg   + (size_t)LSEQ * HV;
    __bf16* hidden_bf = (__bf16*)(bet + (size_t)LSEQ * HV);
    __bf16* wqkvg_bf  = hidden_bf + (size_t)LSEQ * HDIM;          // 3072*2048
    __bf16* wo_bf     = wqkvg_bf + (size_t)PROJ_DIM * HDIM;       // 2048*1024
    __bf16* normed_bf = wo_bf + (size_t)HDIM * VALUE_DIM;         // L*1024

    // 0. one-time f32 -> bf16 conversions for GEMM operands
    f32_to_bf16_kernel<<<(LSEQ * HDIM) / 1024, 256, 0, stream>>>(hidden, hidden_bf);
    f32_to_bf16_kernel<<<(PROJ_DIM * HDIM) / 1024, 256, 0, stream>>>(w_qkvg, wqkvg_bf);
    f32_to_bf16_kernel<<<(HDIM * VALUE_DIM) / 1024, 256, 0, stream>>>(w_o, wo_bf);

    // 1. proj = hidden @ w_qkvg^T     (WMMA bf16, async-LDS pipelined)
    gemm_bf16_wmma<<<dim3(PROJ_DIM / 256, LSEQ / 128), 256, 0, stream>>>(
        hidden_bf, wqkvg_bf, proj, LSEQ, PROJ_DIM, HDIM);
    // 2. a/b projections
    ab_kernel<<<LSEQ, 256, 0, stream>>>(hidden, w_a, w_b, aa, bb);
    // 3. causal conv + SiLU
    conv_silu_kernel<<<(LSEQ * CONV_DIM) / 256, 256, 0, stream>>>(proj, conv_w, conv);
    // 4. q/k L2 norms (in place)
    qknorm_kernel<<<(LSEQ * 2 * HK) / 8, 256, 0, stream>>>(conv);
    // 5. decay / beta gates
    gate_kernel<<<(LSEQ * HV) / 256, 256, 0, stream>>>(aa, bb, A_log, dt_bias, eg, bet);
    // 6. sequential delta-rule scan
    scan_kernel<<<HV, 256, 0, stream>>>(conv, eg, bet, core);
    // 7. RMS-norm + gated SiLU -> bf16
    normgate_kernel<<<(LSEQ * HV) / 8, 256, 0, stream>>>(core, proj, o_nw, normed_bf);
    // 8. out = normed @ w_o^T         (WMMA bf16)
    gemm_bf16_wmma<<<dim3(HDIM / 256, LSEQ / 128), 256, 0, stream>>>(
        normed_bf, wo_bf, out, LSEQ, HDIM, VALUE_DIM);
}
